// SeqEnc_9139690406007
// MI455X (gfx1250) — compile-verified, matrix operand order checked by reading, not run on hardware
//
#include <hip/hip_runtime.h>
#include <hip/hip_bf16.h>
#include <math.h>

// ---------------------------------------------------------------------------
// Types for WMMA fragments (gfx1250, wave32)
// ---------------------------------------------------------------------------
typedef __attribute__((ext_vector_type(16))) _Float16 v16h;
typedef __attribute__((ext_vector_type(8)))  _Float16 v8h;
typedef __attribute__((ext_vector_type(8)))  float    v8f;

// Problem constants
#define BB   64
#define SS   256
#define AA   8
#define BA   512            // B*A
#define HH   256
#define MLPH 512
#define DIN  160            // STATE_DIM + NUM_ACTIONS
#define OUTD 256
#define S_CHUNK 32          // MLP processed in S-chunks of 32 (16384 rows)
#define CHUNK_ROWS (S_CHUNK * BA)

// ---------------------------------------------------------------------------
// WMMA fragment load, per the CDNA5 ISA 16-bit A/B layout:
//   lane L: r = L&15 (row), hi = L>>4
//   v16h elements 0..7  = K[kb + hi*8 .. +8)
//   v16h elements 8..15 = K[kb + 16 + hi*8 .. +8)
// Both operands are stored [rows, K] row-major (K contiguous), so one helper
// serves A and W.  Two 16B loads per lane -> global_load_b128 pairs.
// ---------------------------------------------------------------------------
__device__ __forceinline__ v16h load_frag16(const _Float16* __restrict__ base,
                                            int ld, int row0, int kb) {
  const int lane = threadIdx.x & 31;
  const int r    = lane & 15;
  const int hi   = lane >> 4;
  const _Float16* p = base + (size_t)(row0 + r) * ld + kb + hi * 8;
  union { v16h v; v8h h[2]; } u;
  u.h[0] = *(const v8h*)(p);
  u.h[1] = *(const v8h*)(p + 16);
  return u.v;
}

__device__ __forceinline__ v8f wmma_f16(v16h a, v16h b, v8f c) {
  return __builtin_amdgcn_wmma_f32_16x16x32_f16(
      /*neg_a=*/false, a, /*neg_b=*/false, b,
      /*c_mod=*/(short)0, c, /*reuse_a=*/false, /*reuse_b=*/false);
}

// ---------------------------------------------------------------------------
// Tiled WMMA GEMM: C[M,N] = A[M,K] @ W[N,K]^T (+ A2[M,K2] @ W2[N,K2]^T) + bias
// 128 threads = 4 waves; block tile 128x64; wave tile 32x64
// (2 M-tiles x 4 N-tiles = 8 f32 accumulators per wave).
// K is processed two 32-blocks at a time with all 12 fragment loads issued
// before the first WMMA, so block n+1 loads overlap block n math (partial
// s_wait_loadcnt instead of full stalls).  Requirements: M%128==0, N%64==0,
// K%32==0 -- all shapes here satisfy this.
// ---------------------------------------------------------------------------
template<bool RELU, bool HALF_OUT, bool TRANS_OUT, bool DUAL>
__global__ void __launch_bounds__(128)
wmma_gemm_kernel(const _Float16* __restrict__ A,  const _Float16* __restrict__ W,
                 const _Float16* __restrict__ A2, const _Float16* __restrict__ W2,
                 const float* __restrict__ bias,
                 float* __restrict__ Cf, _Float16* __restrict__ Ch,
                 int M, int N, int K, int K2)
{
  const int wave = threadIdx.x >> 5;
  const int row0 = blockIdx.y * 128 + wave * 32;
  const int col0 = blockIdx.x * 64;

  const v8f vzero = {0.f,0.f,0.f,0.f,0.f,0.f,0.f,0.f};
  v8f acc[2][4];
#pragma unroll
  for (int mi = 0; mi < 2; ++mi)
#pragma unroll
    for (int t = 0; t < 4; ++t) acc[mi][t] = vzero;

  const int lane = threadIdx.x & 31;

  auto kloop = [&](const _Float16* __restrict__ Ap,
                   const _Float16* __restrict__ Wp, int Kl) {
    int kb = 0;
    for (; kb + 64 <= Kl; kb += 64) {
      // ---- issue ALL loads for two K-blocks up front ----
      v16h a0[2], b0[4], a1[2], b1[4];
#pragma unroll
      for (int mi = 0; mi < 2; ++mi) a0[mi] = load_frag16(Ap, Kl, row0 + mi * 16, kb);
#pragma unroll
      for (int t = 0; t < 4; ++t)    b0[t]  = load_frag16(Wp, Kl, col0 + t * 16, kb);
#pragma unroll
      for (int mi = 0; mi < 2; ++mi) a1[mi] = load_frag16(Ap, Kl, row0 + mi * 16, kb + 32);
#pragma unroll
      for (int t = 0; t < 4; ++t)    b1[t]  = load_frag16(Wp, Kl, col0 + t * 16, kb + 32);
      // warm the next chunk into cache (global_prefetch_b8)
      if (kb + 128 <= Kl) {
        __builtin_prefetch(Ap + (size_t)(row0 + (lane & 15)) * Kl + kb + 128, 0, 1);
        __builtin_prefetch(Wp + (size_t)(col0 + (lane & 15)) * Kl + kb + 128, 0, 1);
      }
      // ---- math ----
#pragma unroll
      for (int mi = 0; mi < 2; ++mi)
#pragma unroll
        for (int t = 0; t < 4; ++t) acc[mi][t] = wmma_f16(a0[mi], b0[t], acc[mi][t]);
#pragma unroll
      for (int mi = 0; mi < 2; ++mi)
#pragma unroll
        for (int t = 0; t < 4; ++t) acc[mi][t] = wmma_f16(a1[mi], b1[t], acc[mi][t]);
    }
    if (kb < Kl) {  // remainder K-block of 32 (the K=160 layer)
      v16h a0[2], b0[4];
#pragma unroll
      for (int mi = 0; mi < 2; ++mi) a0[mi] = load_frag16(Ap, Kl, row0 + mi * 16, kb);
#pragma unroll
      for (int t = 0; t < 4; ++t)    b0[t]  = load_frag16(Wp, Kl, col0 + t * 16, kb);
#pragma unroll
      for (int mi = 0; mi < 2; ++mi)
#pragma unroll
        for (int t = 0; t < 4; ++t) acc[mi][t] = wmma_f16(a0[mi], b0[t], acc[mi][t]);
    }
  };

  kloop(A, W, K);
  if (DUAL) kloop(A2, W2, K2);

  // ---- epilogue: C/D layout lane L -> n = L&15, hi = L>>4; elem v -> row v+hi*8
  const int n  = lane & 15;
  const int hi = lane >> 4;
#pragma unroll
  for (int mi = 0; mi < 2; ++mi) {
#pragma unroll
    for (int t = 0; t < 4; ++t) {
      const int col = col0 + t * 16 + n;
      const float bv = bias ? bias[col] : 0.f;
#pragma unroll
      for (int v = 0; v < 8; ++v) {
        const int row = row0 + mi * 16 + hi * 8 + v;
        float x = acc[mi][t][v] + bv;
        if (RELU) x = fmaxf(x, 0.f);
        if (HALF_OUT) {
          if (TRANS_OUT) Ch[(size_t)col * M + row] = (_Float16)x;
          else           Ch[(size_t)row * N + col] = (_Float16)x;
        } else {
          Cf[(size_t)row * N + col] = x;
        }
      }
    }
  }
}

// ---------------------------------------------------------------------------
// Elementwise / utility kernels
// ---------------------------------------------------------------------------
__device__ __forceinline__ float sigmoidf_(float x) { return 1.f / (1.f + __expf(-x)); }

__global__ void f2h_kernel(const float* __restrict__ in, _Float16* __restrict__ out,
                           int n, float mul) {
  int i = blockIdx.x * blockDim.x + threadIdx.x;
  if (i < n) out[i] = (_Float16)(in[i] * mul);
}

__global__ void bias_combine_kernel(const float* __restrict__ a,
                                    const float* __restrict__ b,
                                    float* __restrict__ out, int n) {
  int i = blockIdx.x * blockDim.x + threadIdx.x;
  if (i < n) out[i] = a[i] + b[i];
}

__global__ void init_state_kernel(const float* __restrict__ h0, const float* __restrict__ c0,
                                  float* __restrict__ h, float* __restrict__ c,
                                  _Float16* __restrict__ hh, float* __restrict__ hsum) {
  int i = blockIdx.x * blockDim.x + threadIdx.x;
  if (i < BA * HH) {
    float hv = h0[i];
    h[i] = hv; c[i] = c0[i]; hh[i] = (_Float16)hv; hsum[i] = 0.f;
  }
}

// Build x0[f16] = concat(state, action) for one S-chunk, rows = S_CHUNK*BA.
__global__ void pack_kernel(const float* __restrict__ state,
                            const float* __restrict__ actions,
                            _Float16* __restrict__ x0, int sBase) {
  int idx = blockIdx.x * blockDim.x + threadIdx.x;
  if (idx >= CHUNK_ROWS * DIN) return;
  int rc = idx / DIN, cc = idx - rc * DIN;
  int sLocal = rc >> 9;          // /BA
  int ba = rc & (BA - 1);
  int s = sBase + sLocal;
  int b = ba >> 3, a = ba & 7;
  float val;
  if (cc < 128) val = state[((size_t)b * SS + s) * 128 + cc];
  else          val = actions[(((size_t)b * SS + s) * AA + a) * 32 + (cc - 128)];
  x0[(size_t)rc * DIN + cc] = (_Float16)val;
}

// LSTM cell pointwise: gates[BA,4H] -> new c, pre-attention h (f32 + f16).
__global__ void lstm_pointwise_kernel(const float* __restrict__ gates,
                                      float* __restrict__ c, float* __restrict__ h,
                                      _Float16* __restrict__ hh) {
  int idx = blockIdx.x * blockDim.x + threadIdx.x;
  if (idx >= BA * HH) return;
  int row = idx >> 8, j = idx & 255;
  const float* g = gates + (size_t)row * (4 * HH);
  float iv = sigmoidf_(g[j]);
  float fv = sigmoidf_(g[HH + j]);
  float gv = tanhf(g[2 * HH + j]);
  float ov = sigmoidf_(g[3 * HH + j]);
  float cn = fv * c[idx] + iv * gv;
  c[idx] = cn;
  float hn = ov * tanhf(cn);
  h[idx] = hn;
  hh[idx] = (_Float16)hn;
}

// Row softmax over 512 logits (with scale), f32 in -> f16 probs out.
__global__ void softmax_kernel(const float* __restrict__ scores,
                               _Float16* __restrict__ probs, float scale) {
  __shared__ float red[128];
  const int row = blockIdx.x;
  const float* s = scores + (size_t)row * BA;
  float vals[4];
  float mx = -INFINITY;
#pragma unroll
  for (int i = 0; i < 4; ++i) {
    vals[i] = s[threadIdx.x + i * 128] * scale;
    mx = fmaxf(mx, vals[i]);
  }
  red[threadIdx.x] = mx; __syncthreads();
  for (int off = 64; off; off >>= 1) {
    if (threadIdx.x < off) red[threadIdx.x] = fmaxf(red[threadIdx.x], red[threadIdx.x + off]);
    __syncthreads();
  }
  mx = red[0]; __syncthreads();
  float sum = 0.f;
#pragma unroll
  for (int i = 0; i < 4; ++i) { vals[i] = __expf(vals[i] - mx); sum += vals[i]; }
  red[threadIdx.x] = sum; __syncthreads();
  for (int off = 64; off; off >>= 1) {
    if (threadIdx.x < off) red[threadIdx.x] += red[threadIdx.x + off];
    __syncthreads();
  }
  float inv = 1.f / red[0];
#pragma unroll
  for (int i = 0; i < 4; ++i)
    probs[(size_t)row * BA + threadIdx.x + i * 128] = (_Float16)(vals[i] * inv);
}

// h = h + attout ; hsum += h ; refresh f16 copy for next step's GEMMs.
__global__ void hupdate_kernel(const float* __restrict__ attout,
                               float* __restrict__ h, _Float16* __restrict__ hh,
                               float* __restrict__ hsum) {
  int idx = blockIdx.x * blockDim.x + threadIdx.x;
  if (idx >= BA * HH) return;
  float hn = h[idx] + attout[idx];
  h[idx] = hn;
  hh[idx] = (_Float16)hn;
  hsum[idx] += hn;
}

// ---------------------------------------------------------------------------
// Host side
// ---------------------------------------------------------------------------
static inline void* bump(char*& p, size_t bytes) {
  void* r = (void*)p;
  p += (bytes + 255) & ~(size_t)255;
  return r;
}

extern "C" void kernel_launch(void* const* d_in, const int* in_sizes, int n_in,
                              void* d_out, int out_size, void* d_ws, size_t ws_size,
                              hipStream_t stream) {
  (void)in_sizes; (void)n_in; (void)out_size; (void)ws_size;

  const float* state   = (const float*)d_in[0];
  const float* actions = (const float*)d_in[1];
  const float* h0      = (const float*)d_in[2];
  const float* c0      = (const float*)d_in[3];
  const float* mw[4]   = {(const float*)d_in[4], (const float*)d_in[6],
                          (const float*)d_in[8], (const float*)d_in[10]};
  const float* mb[4]   = {(const float*)d_in[5], (const float*)d_in[7],
                          (const float*)d_in[9], (const float*)d_in[11]};
  const float* w_ih    = (const float*)d_in[12];
  const float* w_hh    = (const float*)d_in[13];
  const float* b_ih    = (const float*)d_in[14];
  const float* b_hh    = (const float*)d_in[15];
  const float* wq      = (const float*)d_in[16];
  const float* bq      = (const float*)d_in[17];
  const float* wk      = (const float*)d_in[18];
  const float* bk      = (const float*)d_in[19];
  const float* wv      = (const float*)d_in[20];
  const float* bv      = (const float*)d_in[21];
  const float* w_out   = (const float*)d_in[22];
  const float* b_out   = (const float*)d_in[23];

  // ---- workspace carve-out (all 256B aligned) ----
  char* p = (char*)d_ws;
  _Float16* w0h   = (_Float16*)bump(p, (size_t)MLPH * DIN  * 2);
  _Float16* w1h   = (_Float16*)bump(p, (size_t)MLPH * MLPH * 2);
  _Float16* w2h   = (_Float16*)bump(p, (size_t)MLPH * MLPH * 2);
  _Float16* w3h   = (_Float16*)bump(p, (size_t)HH   * MLPH * 2);
  _Float16* wihh  = (_Float16*)bump(p, (size_t)4*HH * HH   * 2);
  _Float16* whhh  = (_Float16*)bump(p, (size_t)4*HH * HH   * 2);
  _Float16* wqh   = (_Float16*)bump(p, (size_t)HH * HH * 2);
  _Float16* wkh   = (_Float16*)bump(p, (size_t)HH * HH * 2);
  _Float16* wvh   = (_Float16*)bump(p, (size_t)HH * HH * 2);
  _Float16* wouth = (_Float16*)bump(p, (size_t)OUTD * HH * 2);
  float*    bsum  = (float*)   bump(p, (size_t)4*HH * 4);
  _Float16* xh    = (_Float16*)bump(p, (size_t)SS * BA * HH * 2);   // MLP output, all steps
  _Float16* x0    = (_Float16*)bump(p, (size_t)CHUNK_ROWS * DIN * 2);
  _Float16* bufA  = (_Float16*)bump(p, (size_t)CHUNK_ROWS * MLPH * 2);
  _Float16* bufB  = (_Float16*)bump(p, (size_t)CHUNK_ROWS * MLPH * 2);
  float*    hbuf  = (float*)   bump(p, (size_t)BA * HH * 4);
  float*    cbuf  = (float*)   bump(p, (size_t)BA * HH * 4);
  _Float16* hhalf = (_Float16*)bump(p, (size_t)BA * HH * 2);
  float*    hsum  = (float*)   bump(p, (size_t)BA * HH * 4);
  _Float16* hsumh = (_Float16*)bump(p, (size_t)BA * HH * 2);
  float*    gates = (float*)   bump(p, (size_t)BA * 4*HH * 4);
  _Float16* qh    = (_Float16*)bump(p, (size_t)BA * HH * 2);
  _Float16* kh    = (_Float16*)bump(p, (size_t)BA * HH * 2);
  _Float16* vTh   = (_Float16*)bump(p, (size_t)HH * BA * 2);        // transposed v
  float*    scores= (float*)   bump(p, (size_t)BA * BA * 4);
  float*    attout= (float*)   bump(p, (size_t)BA * HH * 4);
  _Float16* probs = (_Float16*)bump(p, (size_t)BA * BA * 2);

  const dim3 blk(128);
  auto conv = [&](const float* src, _Float16* dst, int n) {
    f2h_kernel<<<(n + 255) / 256, 256, 0, stream>>>(src, dst, n, 1.0f);
  };

  // ---- prologue: weight conversion, bias fusion, state init ----
  conv(mw[0], w0h, MLPH * DIN);
  conv(mw[1], w1h, MLPH * MLPH);
  conv(mw[2], w2h, MLPH * MLPH);
  conv(mw[3], w3h, HH * MLPH);
  conv(w_ih, wihh, 4 * HH * HH);
  conv(w_hh, whhh, 4 * HH * HH);
  conv(wq, wqh, HH * HH);
  conv(wk, wkh, HH * HH);
  conv(wv, wvh, HH * HH);
  conv(w_out, wouth, OUTD * HH);
  bias_combine_kernel<<<4, 256, 0, stream>>>(b_ih, b_hh, bsum, 4 * HH);
  init_state_kernel<<<BA * HH / 256, 256, 0, stream>>>(h0, c0, hbuf, cbuf, hhalf, hsum);

  // ---- phase 1: MLP encoder, chunked over S ----
  for (int ci = 0; ci < SS / S_CHUNK; ++ci) {
    pack_kernel<<<(CHUNK_ROWS * DIN) / 256, 256, 0, stream>>>(state, actions, x0, ci * S_CHUNK);
    wmma_gemm_kernel<true,  true, false, false><<<dim3(MLPH/64, CHUNK_ROWS/128), blk, 0, stream>>>(
        x0,   w0h, nullptr, nullptr, mb[0], nullptr, bufA, CHUNK_ROWS, MLPH, DIN,  0);
    wmma_gemm_kernel<true,  true, false, false><<<dim3(MLPH/64, CHUNK_ROWS/128), blk, 0, stream>>>(
        bufA, w1h, nullptr, nullptr, mb[1], nullptr, bufB, CHUNK_ROWS, MLPH, MLPH, 0);
    wmma_gemm_kernel<true,  true, false, false><<<dim3(MLPH/64, CHUNK_ROWS/128), blk, 0, stream>>>(
        bufB, w2h, nullptr, nullptr, mb[2], nullptr, bufA, CHUNK_ROWS, MLPH, MLPH, 0);
    wmma_gemm_kernel<false, true, false, false><<<dim3(HH/64,   CHUNK_ROWS/128), blk, 0, stream>>>(
        bufA, w3h, nullptr, nullptr, mb[3], nullptr,
        xh + (size_t)ci * CHUNK_ROWS * HH, CHUNK_ROWS, HH, MLPH, 0);
  }

  // ---- phase 2: LSTM + cross-talk attention scan ----
  const float scale = 0.0625f;  // 1/sqrt(256)
  for (int s = 0; s < SS; ++s) {
    const _Float16* xs = xh + (size_t)s * BA * HH;
    // gates = x@w_ih^T + h@w_hh^T + (b_ih+b_hh)
    wmma_gemm_kernel<false, false, false, true><<<dim3(4*HH/64, BA/128), blk, 0, stream>>>(
        xs, wihh, hhalf, whhh, bsum, gates, nullptr, BA, 4 * HH, HH, HH);
    lstm_pointwise_kernel<<<BA * HH / 256, 256, 0, stream>>>(gates, cbuf, hbuf, hhalf);
    // q, k, v(transposed) projections
    wmma_gemm_kernel<false, true, false, false><<<dim3(HH/64, BA/128), blk, 0, stream>>>(
        hhalf, wqh, nullptr, nullptr, bq, nullptr, qh,  BA, HH, HH, 0);
    wmma_gemm_kernel<false, true, false, false><<<dim3(HH/64, BA/128), blk, 0, stream>>>(
        hhalf, wkh, nullptr, nullptr, bk, nullptr, kh,  BA, HH, HH, 0);
    wmma_gemm_kernel<false, true, true,  false><<<dim3(HH/64, BA/128), blk, 0, stream>>>(
        hhalf, wvh, nullptr, nullptr, bv, nullptr, vTh, BA, HH, HH, 0);
    // scores = q @ k^T  (k rows are K-contiguous, same GEMM form)
    wmma_gemm_kernel<false, false, false, false><<<dim3(BA/64, BA/128), blk, 0, stream>>>(
        qh, kh, nullptr, nullptr, nullptr, scores, nullptr, BA, BA, HH, 0);
    softmax_kernel<<<BA, 128, 0, stream>>>(scores, probs, scale);
    // attout = probs @ v   (uses vT as K-contiguous B operand)
    wmma_gemm_kernel<false, false, false, false><<<dim3(HH/64, BA/128), blk, 0, stream>>>(
        probs, vTh, nullptr, nullptr, nullptr, attout, nullptr, BA, HH, BA, 0);
    hupdate_kernel<<<BA * HH / 256, 256, 0, stream>>>(attout, hbuf, hhalf, hsum);
  }

  // ---- phase 3: out = mean_s(h_s) @ w_out^T + b_out (mean folded into f16 conversion) ----
  f2h_kernel<<<BA * HH / 256, 256, 0, stream>>>(hsum, hsumh, BA * HH, 1.0f / (float)SS);
  wmma_gemm_kernel<false, false, false, false><<<dim3(OUTD/64, BA/128), blk, 0, stream>>>(
      hsumh, wouth, nullptr, nullptr, b_out, (float*)d_out, nullptr, BA, OUTD, HH, 0);
}
